// CapsuleModel2_13099650253334
// MI455X (gfx1250) — matrix operand order repaired and checked by reading, not compiled, not run on hardware
//
#include <hip/hip_runtime.h>

// ---------------------------------------------------------------------------
// CapsuleModel2 for MI455X (gfx1250, wave32, WMMA)
//
//  k_fold_small : WV[8][16][32] = w_pos[:, :16] ∘ w_vote ; B3[128] bias fold
//  k_fold_big   : BW bf16, pre-swizzled into WMMA B-fragment layout
//  k_gemm       : G[32768][136] f32 = feature^T @ W3  (bf16 WMMA, f32 acc)
//  k_route1     : 512 blocks = (b,i) x 4 query-chunks; tokens+PE in LDS,
//                 cached-logit softmax, writes p1 (64x16) and log(a1) to ws
//  k_route2     : 128 x 32 threads; 19-class stage -> d_out (8,16,19)
// ---------------------------------------------------------------------------

typedef __attribute__((ext_vector_type(16))) __bf16          v16bf;
typedef __attribute__((ext_vector_type(16))) unsigned short  v16u;
typedef __attribute__((ext_vector_type(8)))  float           v8f;

#define CIN   1280
#define HW    4096
#define NCOL  136      // 128 token dims + 8 act dims
#define NTILES 9       // N padded to 144 = 9 tiles of 16

// ws layout (bytes)
#define G_OFF   0u                              // 32768*136*4 = 17,825,792
#define WF_OFF  17825792u                       // 40*9*32*16*2 = 368,640
#define B3_OFF  18194432u                       // 128*4
#define WV_OFF  18194944u                       // 4096*4
#define P1_OFF  18211328u                       // 128*64*16*4 = 524,288
#define A1L_OFF 18735616u                       // 128*64*4 = 32,768

static __device__ __forceinline__ unsigned short f2bf(float f) {
  unsigned int u = __float_as_uint(f);
  unsigned int r = u + 0x7FFFu + ((u >> 16) & 1u);   // round-to-nearest-even
  return (unsigned short)(r >> 16);
}

static __device__ __forceinline__ float sigmoidf(float x) {
  return 1.0f / (1.0f + __expf(-x));
}

// ---------------------------------------------------------------------------
__global__ __launch_bounds__(256) void k_fold_small(
    const float* __restrict__ b_poses, const float* __restrict__ w_vote,
    const float* __restrict__ b_vote,  const float* __restrict__ w_pos,
    const float* __restrict__ b_pos,
    float* __restrict__ WV, float* __restrict__ B3)
{
  const int tid = threadIdx.x;
  #pragma unroll
  for (int e = 0; e < 16; ++e) {
    int idx = tid + 256 * e;                 // n*512 + v'*32 + d
    int n = idx >> 9, rem = idx & 511, vp = rem >> 5, d = rem & 31;
    float s = 0.f;
    #pragma unroll
    for (int v2 = 0; v2 < 16; ++v2)
      s += w_pos[vp * 18 + v2] * w_vote[(n * 16 + v2) * 32 + d];
    WV[idx] = s;
  }
  if (tid < 128) {
    int n = tid >> 4, vp = tid & 15;
    float s = b_pos[vp];
    for (int v2 = 0; v2 < 16; ++v2) {
      float bv = b_vote[n * 16 + v2];
      for (int d = 0; d < 32; ++d)
        bv += b_poses[n * 32 + d] * w_vote[(n * 16 + v2) * 32 + d];
      s += w_pos[vp * 18 + v2] * bv;
    }
    B3[tid] = s;
  }
}

// ---------------------------------------------------------------------------
// Folded weights in WMMA B-fragment order:
//   BW[((kt*9 + nt)*32 + lane)*16 + i] = W3[k][j] (bf16)
//   k = kt*32 + (lane>>4)*16 + i ;  j = nt*16 + (lane&15)
// ---------------------------------------------------------------------------
__global__ __launch_bounds__(256) void k_fold_big(
    const float* __restrict__ w_poses, const float* __restrict__ w_acts,
    const float* __restrict__ WV, unsigned short* __restrict__ BW)
{
  int idx  = blockIdx.x * 256 + threadIdx.x;   // 0 .. 184319
  int i    = idx & 15;
  int lane = (idx >> 4) & 31;
  int r2   = idx >> 9;                         // kt*9 + nt
  int nt   = r2 % NTILES;
  int kt   = r2 / NTILES;
  int k    = (kt << 5) + ((lane >> 4) << 4) + i;    // 0..1279
  int j    = (nt << 4) + (lane & 15);               // 0..143
  float s = 0.f;
  if (j < 128) {
    int n = j >> 4, vp = j & 15;
    const float* wv = WV + (n * 16 + vp) * 32;
    const float* wp = w_poses + (n * 32) * CIN + k;
    #pragma unroll
    for (int d = 0; d < 32; ++d)
      s += wv[d] * wp[d * CIN];
  } else if (j < NCOL) {
    s = w_acts[(j - 128) * CIN + k];
  }
  BW[idx] = f2bf(s);
}

// ---------------------------------------------------------------------------
// GEMM: block = 128 rows x 144 cols, 8 waves (each: 16 rows x 9 N-tiles).
// ---------------------------------------------------------------------------
__global__ __launch_bounds__(256) void k_gemm(
    const float* __restrict__ feat, const unsigned short* __restrict__ BW,
    float* __restrict__ G)
{
  __shared__ unsigned short Al[8 * 32 * 16];    // 8192 B

  const int tid  = threadIdx.x;
  const int wave = tid >> 5, lane = tid & 31;
  const int m0   = blockIdx.x * 128;
  const int b    = m0 >> 12;                    // blocks never straddle images
  const int hw0  = m0 & (HW - 1);
  const float* fb = feat + (size_t)b * CIN * HW + hw0;
  const v16u* Bf = (const v16u*)BW;             // [ (kt*9+nt)*32 + lane ]

  v8f acc[NTILES];
  #pragma unroll
  for (int nt = 0; nt < NTILES; ++nt)
    #pragma unroll
    for (int i = 0; i < 8; ++i) acc[nt][i] = 0.f;

  const v16u* Af = (const v16u*)&Al[((wave << 5) + lane) << 4];

  for (int k0 = 0; k0 < CIN; k0 += 32) {
    const int ktIdx = k0 >> 5;

    // stage A: 128 m x 32 k, two consecutive k packed per b32 LDS store
    #pragma unroll
    for (int e = 0; e < 8; ++e) {
      int idx = tid + 256 * e;                  // 0..2047
      int kp  = idx >> 7;                       // 0..15
      int m   = idx & 127;
      int k   = kp << 1;
      float f0 = fb[(k0 + k) * HW + m];
      float f1 = fb[(k0 + k + 1) * HW + m];
      unsigned pack = (unsigned)f2bf(f0) | ((unsigned)f2bf(f1) << 16);
      int u  = k >> 4;
      int i  = k & 15;                          // even
      int mt = m >> 4, mr = m & 15;
      *(unsigned*)&Al[((((mt << 5) + (u << 4) + mr) << 4) + i)] = pack;
    }

    // prefetch next K-tile (feature rows + swizzled weights)
    if (k0 + 32 < CIN) {
      int kk = tid >> 3, off = (tid & 7) << 4;
      __builtin_prefetch(&fb[(k0 + 32 + kk) * HW + off], 0, 3);
      if (tid < 144)
        __builtin_prefetch((const char*)BW + (size_t)(ktIdx + 1) * 9216 + tid * 64, 0, 3);
    }

    __syncthreads();

    v16u au = *Af;                              // 32B contiguous per lane
    v16bf a = __builtin_bit_cast(v16bf, au);

    #pragma unroll
    for (int nt = 0; nt < NTILES; ++nt) {
      v16u bu = Bf[(ktIdx * NTILES + nt) * 32 + lane];   // 32B contiguous per lane
      v16bf bb = __builtin_bit_cast(v16bf, bu);
      acc[nt] = __builtin_amdgcn_wmma_f32_16x16x32_bf16(
          false, a, false, bb, (short)0, acc[nt], false, false);
    }
    __syncthreads();
  }

  // epilogue: C layout VGPR r -> row = r + 8*(lane>>4), col = lane&15
  const int rowbase = m0 + wave * 16 + ((lane >> 4) << 3);
  #pragma unroll
  for (int nt = 0; nt < NTILES; ++nt) {
    int col = nt * 16 + (lane & 15);
    if (col < NCOL) {
      #pragma unroll
      for (int r = 0; r < 8; ++r)
        G[(rowbase + r) * NCOL + col] = acc[nt][r];
    }
  }
}

// ---------------------------------------------------------------------------
// Routing stage 1: block = (b,i) x query-chunk (16 queries).
// Tokens rebuilt per block (G is L2-hot); per-wave LDS logit cache makes
// the softmax single-dot-pass. Writes p1 rows and log(a1+1e-6) to ws.
// ---------------------------------------------------------------------------
__global__ __launch_bounds__(256) void k_route1(
    const float* __restrict__ G, const int* __restrict__ coords,
    const unsigned char* __restrict__ mask,
    const float* __restrict__ B3, const float* __restrict__ b_acts,
    const float* __restrict__ w_pos,
    const float* __restrict__ Q1, const float* __restrict__ Wv1,
    const float* __restrict__ wact1, const float* __restrict__ bact1,
    float* __restrict__ P1, float* __restrict__ A1L)
{
  extern __shared__ float sm[];
  float* tok = sm;                  // [2048][17] : 16 dims + masked log-act

  const int tid = threadIdx.x;
  const int bi  = blockIdx.x >> 2;        // b*16 + i
  const int q0  = (blockIdx.x & 3) << 4;  // query chunk base
  const int b   = bi >> 4;
  const int* cy = coords + bi * 512;
  const int* cx = cy + 256;
  const unsigned char* mk = mask + bi * 256;

  // build tokens
  #pragma unroll
  for (int e = 0; e < 8; ++e) {
    int t = tid + 256 * e;          // n*256 + p
    int n = t >> 8, p = t & 255;
    int y = cy[p], x = cx[p];
    float mean = 0.5f * (float)(y + x);
    float yr = ((float)y - mean) * (1.f / 64.f);
    float xr = ((float)x - mean) * (1.f / 64.f);
    const float* g = G + ((b << 12) + (y << 6) + x) * NCOL;
    float* tv = tok + t * 17;
    #pragma unroll
    for (int v = 0; v < 16; ++v)
      tv[v] = g[n * 16 + v] + B3[n * 16 + v] +
              yr * w_pos[v * 18 + 16] + xr * w_pos[v * 18 + 17];
    float act = sigmoidf(g[128 + n] + b_acts[n]);
    tv[16] = mk[p] ? logf(act + 1e-6f) : -1.0e9f;
  }
  __syncthreads();

  const int wv = tid >> 5, lane = tid & 31;
  float* lc = sm + 2048 * 17 + (wv << 11);   // per-wave logit cache [2048]

  for (int j = 0; j < 2; ++j) {
    const int q = q0 + (wv << 1) + j;
    float qv[16];
    #pragma unroll
    for (int v = 0; v < 16; ++v) qv[v] = Q1[q * 16 + v];

    // pass 1: logits -> cache, running max
    float mx = -3.4e38f;
    for (int t = lane; t < 2048; t += 32) {
      const float* tv = tok + t * 17;
      float s = 0.f;
      #pragma unroll
      for (int v = 0; v < 16; ++v) s += qv[v] * tv[v];
      s = s * 0.25f + tv[16];
      lc[t] = s;
      mx = fmaxf(mx, s);
    }
    #pragma unroll
    for (int o = 16; o > 0; o >>= 1) mx = fmaxf(mx, __shfl_xor(mx, o, 32));

    // pass 2: exp from cache, weighted token sum
    float se = 0.f, accv[16];
    #pragma unroll
    for (int v = 0; v < 16; ++v) accv[v] = 0.f;
    for (int t = lane; t < 2048; t += 32) {
      const float* tv = tok + t * 17;
      float w = __expf(lc[t] - mx);
      se += w;
      #pragma unroll
      for (int v = 0; v < 16; ++v) accv[v] += w * tv[v];
    }
    #pragma unroll
    for (int o = 16; o > 0; o >>= 1) {
      se += __shfl_xor(se, o, 32);
      #pragma unroll
      for (int v = 0; v < 16; ++v) accv[v] += __shfl_xor(accv[v], o, 32);
    }

    float r = 0.f;
    if (lane < 16) {
      float pv = 0.f;
      #pragma unroll
      for (int v = 0; v < 16; ++v) pv += accv[v] * Wv1[v * 16 + lane];
      pv /= se;
      P1[(bi << 10) + (q << 4) + lane] = pv;
      r = pv * wact1[lane];
    }
    #pragma unroll
    for (int o = 16; o > 0; o >>= 1) r += __shfl_xor(r, o, 32);
    if (lane == 0)
      A1L[(bi << 6) + q] = logf(sigmoidf(r + bact1[0]) + 1e-6f);
  }
}

// ---------------------------------------------------------------------------
// Routing stage 2: one wave per (b,i); 19 classes over 64 p1 tokens.
// ---------------------------------------------------------------------------
__global__ __launch_bounds__(32) void k_route2(
    const float* __restrict__ P1, const float* __restrict__ A1L,
    const float* __restrict__ Q2, const float* __restrict__ wact2,
    const float* __restrict__ bact2, float* __restrict__ out)
{
  __shared__ float sp1[64 * 16];
  __shared__ float sa[64];
  const int bi = blockIdx.x;
  const int t  = threadIdx.x;

  #pragma unroll
  for (int e = 0; e < 32; ++e) sp1[t + 32 * e] = P1[(bi << 10) + t + 32 * e];
  sa[t]      = A1L[(bi << 6) + t];
  sa[t + 32] = A1L[(bi << 6) + t + 32];
  __syncthreads();

  if (t < 19) {
    float q2[16];
    #pragma unroll
    for (int v = 0; v < 16; ++v) q2[v] = Q2[t * 16 + v];
    float mx = -3.4e38f;
    for (int q = 0; q < 64; ++q) {
      const float* pv = sp1 + q * 16;
      float s = 0.f;
      #pragma unroll
      for (int v = 0; v < 16; ++v) s += q2[v] * pv[v];
      s = s * 0.25f + sa[q];
      mx = fmaxf(mx, s);
    }
    float se = 0.f, ov[16];
    #pragma unroll
    for (int v = 0; v < 16; ++v) ov[v] = 0.f;
    for (int q = 0; q < 64; ++q) {
      const float* pv = sp1 + q * 16;
      float s = 0.f;
      #pragma unroll
      for (int v = 0; v < 16; ++v) s += q2[v] * pv[v];
      s = s * 0.25f + sa[q];
      float w = __expf(s - mx);
      se += w;
      #pragma unroll
      for (int v = 0; v < 16; ++v) ov[v] += w * pv[v];
    }
    float r = 0.f;
    #pragma unroll
    for (int v = 0; v < 16; ++v) r += (ov[v] / se) * wact2[v];
    out[bi * 19 + t] = sigmoidf(r + bact2[0]);
  }
}

// ---------------------------------------------------------------------------
extern "C" void kernel_launch(void* const* d_in, const int* in_sizes, int n_in,
                              void* d_out, int out_size, void* d_ws, size_t ws_size,
                              hipStream_t stream)
{
  (void)in_sizes; (void)n_in; (void)out_size; (void)ws_size;
  const float* feat           = (const float*)d_in[0];
  const int* coords           = (const int*)d_in[1];
  const unsigned char* mask   = (const unsigned char*)d_in[2];
  const float* w_poses        = (const float*)d_in[3];
  const float* b_poses        = (const float*)d_in[4];
  const float* w_acts         = (const float*)d_in[5];
  const float* b_acts         = (const float*)d_in[6];
  const float* w_vote         = (const float*)d_in[7];
  const float* b_vote         = (const float*)d_in[8];
  const float* w_pos          = (const float*)d_in[9];
  const float* b_pos          = (const float*)d_in[10];
  const float* Q1             = (const float*)d_in[11];
  const float* Wv1            = (const float*)d_in[12];
  const float* wact1          = (const float*)d_in[13];
  const float* bact1          = (const float*)d_in[14];
  const float* Q2             = (const float*)d_in[15];
  const float* wact2          = (const float*)d_in[16];
  const float* bact2          = (const float*)d_in[17];

  char* ws = (char*)d_ws;
  float*          G   = (float*)(ws + G_OFF);
  unsigned short* BW  = (unsigned short*)(ws + WF_OFF);
  float*          B3  = (float*)(ws + B3_OFF);
  float*          WV  = (float*)(ws + WV_OFF);
  float*          P1  = (float*)(ws + P1_OFF);
  float*          A1L = (float*)(ws + A1L_OFF);

  k_fold_small<<<1, 256, 0, stream>>>(b_poses, w_vote, b_vote, w_pos, b_pos, WV, B3);
  k_fold_big<<<720, 256, 0, stream>>>(w_poses, w_acts, WV, BW);
  k_gemm<<<256, 256, 0, stream>>>(feat, BW, G);

  size_t smem1 = (size_t)(2048 * 17 + 8 * 2048) * sizeof(float);   // 204,800 B
  k_route1<<<512, 256, smem1, stream>>>(G, coords, mask, B3, b_acts, w_pos,
                                        Q1, Wv1, wact1, bact1, P1, A1L);
  k_route2<<<128, 32, 0, stream>>>(P1, A1L, Q2, wact2, bact2, (float*)d_out);
}
